// SigCubicalEcc_22247930593540
// MI455X (gfx1250) — compile-verified, matrix operand order checked by reading, not run on hardware
//
#include <hip/hip_runtime.h>

typedef __attribute__((ext_vector_type(16))) _Float16 v16h;
typedef __attribute__((ext_vector_type(8)))  float    v8f;

#define B_      64
#define C_      3
#define H_      64
#define W_      64
#define STEPS_  32
#define LAM_    200.0f
#define T_MIN_  0.02f
#define T_MAX_  0.28f
#define BIG_    1.0e9f

// Cell counts for the T-construction cubical complex over a 64x64 image
#define NV_   (65 * 65)                 // vertices  : 4225  (sign +)
#define NEH_  (65 * 64)                 // h-edges   : 4160  (sign -)
#define NEV_  (64 * 65)                 // v-edges   : 4160  (sign -)
#define NF_   (64 * 64)                 // faces     : 4096  (sign +)
#define NCELLS_     (NV_ + NEH_ + NEV_ + NF_)   // 16641
#define NGROUPS_    ((NCELLS_ + 31) / 32)       // 521
#define NCELLS_PAD_ (NGROUPS_ * 32)             // 16672 (31 pad cells)
#define NWAVES_     8
#define NTHREADS_   256

// Padded pixel read: outside the image the pad value is BIG (so boundary
// cells take the single adjacent pixel under min), matching jnp.pad(...BIG).
__device__ __forceinline__ float px(const float* __restrict__ xi, int r, int c) {
    return ((unsigned)r < (unsigned)H_ && (unsigned)c < (unsigned)W_)
               ? xi[r * W_ + c]
               : BIG_;
}

__global__ __launch_bounds__(NTHREADS_) void ecc_wmma_kernel(
    const float* __restrict__ x, float* __restrict__ out) {
    extern __shared__ float smem[];     // [NCELLS_PAD_] cell values (pre-mul K2) + [NWAVES_*32] partials
    float* cells = smem;
    float* red   = smem + NCELLS_PAD_;

    const int   bc  = blockIdx.x;                     // flattened (b, channel)
    const int   tid = threadIdx.x;
    const float* xi = x + (size_t)bc * (H_ * W_);

    const float K2 = LAM_ * 1.4426950408889634f;      // LAM * log2(e)
    const float DT = (T_MAX_ - T_MIN_) / (float)(STEPS_ - 1);

    // -------- Stage 1: all cell filtration values into LDS (scaled by K2) ----
    for (int id = tid; id < NCELLS_PAD_; id += NTHREADS_) {
        float f;
        if (id >= NCELLS_) {
            f = BIG_;                                  // pad: sigma == 0 exactly
        } else if (id < NV_) {                         // vertex grid 65x65
            int r = id / 65, c = id % 65;
            float m0 = fminf(px(xi, r - 1, c - 1), px(xi, r - 1, c));
            float m1 = fminf(px(xi, r,     c - 1), px(xi, r,     c));
            f = fminf(m0, m1);
        } else if (id < NV_ + NEH_) {                  // horizontal edges 65x64
            int id2 = id - NV_;
            int r = id2 / 64, c = id2 % 64;
            f = fminf(px(xi, r - 1, c), px(xi, r, c));
        } else if (id < NV_ + NEH_ + NEV_) {           // vertical edges 64x65
            int id3 = id - (NV_ + NEH_);
            int r = id3 / 65, c = id3 % 65;
            f = fminf(px(xi, r, c - 1), px(xi, r, c));
        } else {                                       // faces = pixels 64x64
            f = xi[id - (NV_ + NEH_ + NEV_)];
        }
        cells[id] = f * K2;
    }
    __syncthreads();

    // -------- Stage 2: per-wave WMMA reduction over cell groups --------------
    const int lane = tid & 31;
    // Wave index as a true SGPR: the group loop becomes a uniform scalar loop,
    // guaranteeing EXEC == all-1s at every WMMA (ISA 7.12 restriction) and
    // keeping loop control off the VALU.
    const int wavS = __builtin_amdgcn_readfirstlane(tid >> 5);
    const int half = lane >> 4;        // which 16 cells of the group this lane covers
    const int ncol = lane & 15;        // B-matrix column == threshold index (mod 16)

    // sigmoid(LAM*(t-f)) = 1 / (1 + 2^(K2*f - K2*t))
    const float cK_lo = K2 * (T_MIN_ + (float)ncol * DT);
    // e(t + 16*DT) = e(t) * 2^(-K2*16*DT): one exp2 serves both t-halves
    const float E16 = __builtin_amdgcn_exp2f(-K2 * 16.0f * DT);

    v16h a_ones;
#pragma unroll
    for (int j = 0; j < 16; ++j) a_ones[j] = (_Float16)1.0f;

    v8f acc_lo = {};
    v8f acc_hi = {};

    for (int g = wavS; g < NGROUPS_; g += NWAVES_) {
        const int base = g * 32 + half * 16;
        v16h b_lo, b_hi;
#pragma unroll
        for (int j = 0; j < 16; ++j) {
            const int   idx = base + j;
            const float a   = cells[idx];   // LDS broadcast within each lane half
            // edges get negative sign in the Euler curve
            const bool  neg = ((unsigned)(idx - NV_)) < (unsigned)(NEH_ + NEV_);
            const float e0  = __builtin_amdgcn_exp2f(a - cK_lo);
            const float e1  = e0 * E16;
            // store (sigma - 0.5) in f16: halves magnitude; exact for pad cells
            float s0 = __builtin_amdgcn_rcpf(1.0f + e0) - 0.5f;
            float s1 = __builtin_amdgcn_rcpf(1.0f + e1) - 0.5f;
            s0 = neg ? -s0 : s0;
            s1 = neg ? -s1 : s1;
            b_lo[j] = (_Float16)s0;
            b_hi[j] = (_Float16)s1;
        }
        // D[m,n] += sum_k B[k,n]  (A = ones): per-threshold reduction over 32 cells
        acc_lo = __builtin_amdgcn_wmma_f32_16x16x32_f16(
            false, a_ones, false, b_lo, (short)0, acc_lo, false, false);
        acc_hi = __builtin_amdgcn_wmma_f32_16x16x32_f16(
            false, a_ones, false, b_hi, (short)0, acc_hi, false, false);
    }

    // -------- Stage 3: cross-wave reduction ---------------------------------
    // D VGPR0, lanes 0..15: row M=0, column N=lane -> partial sum for t_N.
    if (lane < 16) {
        red[wavS * 32 + ncol]      = acc_lo[0];
        red[wavS * 32 + 16 + ncol] = acc_hi[0];
    }
    __syncthreads();

    if (tid < STEPS_) {
        float s = 0.0f;
#pragma unroll
        for (int w2 = 0; w2 < NWAVES_; ++w2) s += red[w2 * 32 + tid];
        // (sigma-0.5) shift correction: 0.5*(V - EH - EV + F) = 0.5*1, plus
        // 31 pad cells at sigma=0 contributing exactly -0.5 each -> +16.0 total.
        out[bc * STEPS_ + tid] = s + 16.0f;
    }
}

extern "C" void kernel_launch(void* const* d_in, const int* in_sizes, int n_in,
                              void* d_out, int out_size, void* d_ws, size_t ws_size,
                              hipStream_t stream) {
    const float* x   = (const float*)d_in[0];
    float*       out = (float*)d_out;
    const size_t shmem = (size_t)(NCELLS_PAD_ + NWAVES_ * 32) * sizeof(float);
    ecc_wmma_kernel<<<B_ * C_, NTHREADS_, shmem, stream>>>(x, out);
}
